// DeepseekMha_88235808129258
// MI455X (gfx1250) — compile-verified, matrix operand order checked by reading, not run on hardware
//
#include <hip/hip_runtime.h>
#include <hip/hip_bf16.h>

#define B_      2
#define S_      2048
#define HIDDEN_ 2048
#define NHEADS  16
#define NKV     4
#define HDIM    128
#define QSIZE   2048
#define KVSIZE  512
#define QKV_N   3072

typedef __bf16 bf16;
typedef bf16  bf16x16 __attribute__((ext_vector_type(16)));
typedef float f32x8   __attribute__((ext_vector_type(8)));
typedef unsigned int u32x4 __attribute__((ext_vector_type(4)));
typedef int   i32x4   __attribute__((ext_vector_type(4)));
typedef int   i32x8   __attribute__((ext_vector_type(8)));

// Async-LDS builtin wants pointers to v4i32 (gcc vector_size style), AS1 src / AS3 dst.
typedef int v4i32 __attribute__((vector_size(16)));
typedef __attribute__((address_space(1))) v4i32* as1_v4p;
typedef __attribute__((address_space(3))) v4i32* as3_v4p;

#if __has_builtin(__builtin_amdgcn_tensor_load_to_lds)
#define USE_TDM 1
#else
#define USE_TDM 0
#endif
#if __has_builtin(__builtin_amdgcn_global_load_async_to_lds_b128)
#define USE_ASYNC 1
#else
#define USE_ASYNC 0
#endif

__device__ __forceinline__ void wait_async0() {
#if USE_ASYNC
#if __has_builtin(__builtin_amdgcn_s_wait_asynccnt)
  __builtin_amdgcn_s_wait_asynccnt(0);
#else
  asm volatile("s_wait_asynccnt 0x0" ::: "memory");
#endif
#endif
}

__device__ __forceinline__ void wait_tensor0() {
#if __has_builtin(__builtin_amdgcn_s_wait_tensorcnt)
  __builtin_amdgcn_s_wait_tensorcnt((short)0);
#else
  asm volatile("s_wait_tensorcnt 0x0" ::: "memory");
#endif
}

#if USE_TDM
// Issue a TDM 2D tile load: tile_x contiguous elements per row, tile_y rows,
// row stride stride0 (elements), into LDS at lds_off. D# per ISA 08_async_tensor §8.
__device__ __forceinline__ void tdm_load_2d(unsigned lds_off, const void* gsrc,
                                            unsigned dsz_log2, unsigned tile_x,
                                            unsigned tile_y,
                                            unsigned long long stride0) {
  unsigned long long ga = (unsigned long long)(size_t)gsrc;
  u32x4 g0;
  g0[0] = 1u;                                             // count=1 (valid user D#)
  g0[1] = lds_off;                                        // lds_addr
  g0[2] = (unsigned)ga;                                   // global_addr[31:0]
  g0[3] = (unsigned)((ga >> 32) & 0x01ffffffu) | (2u << 30); // addr[56:32] | type=2
  unsigned long long td0 = tile_x, td1 = tile_y;
  // group1: data_size@16, tensor_dim0@48, tensor_dim1@80, tile_dim0@112,
  //         tile_dim1@128, tensor_dim0_stride@160
  unsigned long long q0 = ((unsigned long long)dsz_log2 << 16) | (td0 << 48);
  unsigned long long q1 = (td0 >> 16) | (td1 << 16) |
                          ((unsigned long long)(tile_x & 0xffffu) << 48);
  unsigned long long q2 = (unsigned long long)(tile_y & 0xffffu) |
                          ((stride0 & 0xffffffffull) << 32);
  unsigned long long q3 = (stride0 >> 32) & 0xffffull;
  i32x8 g1;
  g1[0] = (int)q0; g1[1] = (int)(q0 >> 32);
  g1[2] = (int)q1; g1[3] = (int)(q1 >> 32);
  g1[4] = (int)q2; g1[5] = (int)(q2 >> 32);
  g1[6] = (int)q3; g1[7] = (int)(q3 >> 32);
  i32x4 gz4 = {0, 0, 0, 0};
  i32x8 gz8 = {0, 0, 0, 0, 0, 0, 0, 0};
  __builtin_amdgcn_tensor_load_to_lds(g0, g1, gz4, gz4, gz8, 0);
}
#endif

__device__ __forceinline__ f32x8 wmma_bf16(bf16x16 a, bf16x16 b, f32x8 c) {
  return __builtin_amdgcn_wmma_f32_16x16x32_bf16(false, a, false, b, (short)0, c, false, false);
}

// A-layout fragment (16x32, MxK) from row-major T; also K^T B-fragments.
template <typename T>
__device__ __forceinline__ bf16x16 frag_row(const T* p, int ld) {
  const int lane = threadIdx.x & 31;
  const T* r = p + (size_t)(lane & 15) * ld + ((lane >> 4) << 3);
  bf16x16 f;
#pragma unroll
  for (int i = 0; i < 8; ++i) f[i] = (bf16)(float)r[i];
#pragma unroll
  for (int i = 0; i < 8; ++i) f[8 + i] = (bf16)(float)r[16 + i];
  return f;
}

// B-layout fragment (32x16, KxN) from row-major [K][N] T.
template <typename T>
__device__ __forceinline__ bf16x16 frag_col(const T* p, int ld) {
  const int lane = threadIdx.x & 31;
  const T* c0 = p + (lane & 15) + (size_t)((lane >> 4) << 3) * ld;
  bf16x16 f;
#pragma unroll
  for (int i = 0; i < 8; ++i) f[i] = (bf16)(float)c0[(size_t)i * ld];
#pragma unroll
  for (int i = 0; i < 8; ++i) f[8 + i] = (bf16)(float)c0[(size_t)(16 + i) * ld];
  return f;
}

// Stage a ROWSxCOLS tile of T into LDS: async b128 copies if available, else sync.
template <typename T, int ROWS, int COLS>
__device__ __forceinline__ void stage_tile(T* lds, const T* g, int ld) {
  constexpr int EPC = 16 / (int)sizeof(T);     // elems per 16B chunk
  constexpr int CPR = COLS / EPC;
  constexpr int TOT = ROWS * CPR;
  const int tid = threadIdx.x;
#pragma unroll
  for (int i = 0; i < TOT / 128; ++i) {
    const int ci = tid + i * 128;
    const int r = ci / CPR, c = (ci % CPR) * EPC;
    const T* gp = g + (size_t)r * ld + c;
    T* lp = lds + r * COLS + c;
#if USE_ASYNC
    __builtin_amdgcn_global_load_async_to_lds_b128(
        (as1_v4p)(void*)const_cast<T*>(gp), (as3_v4p)(void*)lp, 0, 0);
#else
    *(uint4*)lp = *(const uint4*)gp;
#endif
  }
}

// ---------------- GEMM: C[M,N] = A[M,K] * B[K,N], bf16 WMMA, f32 accum ----------------
template <typename TA, typename TC>
__global__ __launch_bounds__(128) void gemm_bf16_kernel(
    const TA* __restrict__ A, const float* __restrict__ Bw, TC* __restrict__ C,
    int K, int lda, int ldb, int ldc) {
  __shared__ __align__(16) TA    As[2][64 * 32];
  __shared__ __align__(16) float Bs[2][32 * 64];

  const int tid  = threadIdx.x;
  const int lane = tid & 31;
  const int wid  = tid >> 5;
  const int m0g  = blockIdx.y * 64;
  const int n0g  = blockIdx.x * 64;

  f32x8 acc0 = {}, acc1 = {}, acc2 = {}, acc3 = {};

  stage_tile<TA, 64, 32>(As[0], A + (size_t)m0g * lda, lda);
  stage_tile<float, 32, 64>(Bs[0], Bw + n0g, ldb);

  for (int k0 = 0; k0 < K; k0 += 32) {
    const int cur = (k0 >> 5) & 1;
    wait_async0();
    __syncthreads();
    if (k0 + 32 < K) {      // prefetch next tile into the other buffer (overlaps compute)
      stage_tile<TA, 64, 32>(As[cur ^ 1], A + (size_t)m0g * lda + (k0 + 32), lda);
      stage_tile<float, 32, 64>(Bs[cur ^ 1], Bw + (size_t)(k0 + 32) * ldb + n0g, ldb);
    }
    bf16x16 a  = frag_row<TA>(&As[cur][wid * 16 * 32], 32);
    bf16x16 b0 = frag_col<float>(&Bs[cur][0],  64);
    bf16x16 b1 = frag_col<float>(&Bs[cur][16], 64);
    bf16x16 b2 = frag_col<float>(&Bs[cur][32], 64);
    bf16x16 b3 = frag_col<float>(&Bs[cur][48], 64);
    acc0 = wmma_bf16(a, b0, acc0);
    acc1 = wmma_bf16(a, b1, acc1);
    acc2 = wmma_bf16(a, b2, acc2);
    acc3 = wmma_bf16(a, b3, acc3);
  }

  const int m = m0g + wid * 16 + ((lane >> 4) << 3);
  const int n = n0g + (lane & 15);
#pragma unroll
  for (int r = 0; r < 8; ++r) {
    C[(size_t)(m + r) * ldc + n +  0] = (TC)acc0[r];
    C[(size_t)(m + r) * ldc + n + 16] = (TC)acc1[r];
    C[(size_t)(m + r) * ldc + n + 32] = (TC)acc2[r];
    C[(size_t)(m + r) * ldc + n + 48] = (TC)acc3[r];
  }
}

// ---------------- RoPE (neox rotate-half) in-place on Q and K columns of qkv ----------------
__global__ void rope_kernel(bf16* __restrict__ qkv, const int* __restrict__ positions) {
  const int t = blockIdx.x * blockDim.x + threadIdx.x;
  const int total = B_ * S_ * (NHEADS + NKV) * (HDIM / 2);
  if (t >= total) return;
  const int i    = t & 63;
  const int head = (t >> 6) % (NHEADS + NKV);
  const int row  = t / (64 * (NHEADS + NKV));
  const int colbase = (head < NHEADS) ? head * HDIM : QSIZE + (head - NHEADS) * HDIM;
  bf16* p = qkv + (size_t)row * QKV_N + colbase;
  const float pos  = (float)positions[row];
  const float freq = __expf(-(float)i * (9.210340371976184f / 64.0f)); // 10000^(-i/64)
  float s, c;
  __sincosf(pos * freq, &s, &c);
  const float x1 = (float)p[i], x2 = (float)p[i + 64];
  p[i]      = (bf16)(x1 * c - x2 * s);
  p[i + 64] = (bf16)(x2 * c + x1 * s);
}

// ---------------- Causal flash attention ----------------
// Block = 4 waves = 4 q-tiles of the SAME head; K/V tiles shared via TDM-staged
// double-buffered LDS (TENSORcnt), Q held in registers, softmax stats in registers.
__global__ __launch_bounds__(128) void flash_attn_kernel(
    const bf16* __restrict__ qkv, bf16* __restrict__ attn) {
  __shared__ __align__(16) bf16 Kb[2][32 * HDIM];
  __shared__ __align__(16) bf16 Vb[2][32 * HDIM];
  __shared__ __align__(16) bf16 Pb[4][16 * 32];

  const int tid  = threadIdx.x;
  const int lane = tid & 31;
  const int wid  = tid >> 5;
  const int h    = blockIdx.y;
  const int b    = blockIdx.z;
  const int q0   = (blockIdx.x * 4 + wid) * 16;
  const int kvh  = h >> 2;

  const bf16* qp = qkv + ((size_t)(b * S_) + q0) * QKV_N + h * HDIM;
  const bf16* kp = qkv + (size_t)(b * S_) * QKV_N + QSIZE + kvh * HDIM;
  const bf16* vp = kp + KVSIZE;

  bf16x16 qa[4];
#pragma unroll
  for (int c = 0; c < 4; ++c) qa[c] = frag_row<bf16>(qp + c * 32, QKV_N);

  float mrow[8], lrow[8];
  f32x8 o[8];
  const f32x8 zero = {};
#pragma unroll
  for (int r = 0; r < 8; ++r) { mrow[r] = -1e30f; lrow[r] = 0.0f; }
#pragma unroll
  for (int c = 0; c < 8; ++c) o[c] = zero;

  const int   hi8   = (lane >> 4) << 3;
  const int   nl    = lane & 15;
  const float scale = 0.08838834764831845f;   // 1/sqrt(128)
  bf16* Pw = Pb[wid];

  const int ktend = 2 * blockIdx.x + 2;       // uniform across block (barriers inside loop)

#if USE_TDM
  if (wid == 0) {
    tdm_load_2d((unsigned)(size_t)&Kb[0][0], kp, 1, HDIM, 32, QKV_N);
    tdm_load_2d((unsigned)(size_t)&Vb[0][0], vp, 1, HDIM, 32, QKV_N);
  }
#endif

  for (int kt = 0; kt < ktend; ++kt) {
    const int cur   = kt & 1;
    const int kbase = kt * 32;
#if USE_TDM
    if (wid == 0) wait_tensor0();
    __syncthreads();                          // K/V[cur] visible to all waves
    if (wid == 0 && kt + 1 < ktend) {         // overlap next tile with compute
      tdm_load_2d((unsigned)(size_t)&Kb[cur ^ 1][0],
                  kp + (size_t)(kbase + 32) * QKV_N, 1, HDIM, 32, QKV_N);
      tdm_load_2d((unsigned)(size_t)&Vb[cur ^ 1][0],
                  vp + (size_t)(kbase + 32) * QKV_N, 1, HDIM, 32, QKV_N);
    }
#else
    __syncthreads();
#pragma unroll
    for (int i = 0; i < 4; ++i) {             // cooperative 32x128 bf16 tile copies
      const int ci = tid + i * 128;
      const int r = ci >> 4, c = (ci & 15) << 3;
      *(uint4*)&Kb[cur][r * HDIM + c] = *(const uint4*)(kp + (size_t)(kbase + r) * QKV_N + c);
      *(uint4*)&Vb[cur][r * HDIM + c] = *(const uint4*)(vp + (size_t)(kbase + r) * QKV_N + c);
    }
    __syncthreads();
#endif

    // S = Q x K^T : two 16x16 tiles over 32 keys, head dim in 4 chunks of 32
    f32x8 s0 = zero, s1 = zero;
#pragma unroll
    for (int c = 0; c < 4; ++c) {
      bf16x16 kb0 = frag_row<bf16>(&Kb[cur][c * 32], HDIM);
      bf16x16 kb1 = frag_row<bf16>(&Kb[cur][16 * HDIM + c * 32], HDIM);
      s0 = wmma_bf16(qa[c], kb0, s0);
      s1 = wmma_bf16(qa[c], kb1, s1);
    }

    // scale + causal mask + online softmax (row stats via half-wave shfl reductions)
#pragma unroll
    for (int r = 0; r < 8; ++r) {
      const int m = q0 + hi8 + r;
      float v0 = (kbase + nl      <= m) ? s0[r] * scale : -1e30f;
      float v1 = (kbase + 16 + nl <= m) ? s1[r] * scale : -1e30f;
      float tm = fmaxf(v0, v1);
      tm = fmaxf(tm, __shfl_xor(tm, 1, 32));
      tm = fmaxf(tm, __shfl_xor(tm, 2, 32));
      tm = fmaxf(tm, __shfl_xor(tm, 4, 32));
      tm = fmaxf(tm, __shfl_xor(tm, 8, 32));
      const float mnew  = fmaxf(mrow[r], tm);
      const float alpha = __expf(mrow[r] - mnew);
      mrow[r] = mnew;
      const float p0 = __expf(v0 - mnew);
      const float p1 = __expf(v1 - mnew);
      float rs = p0 + p1;
      rs += __shfl_xor(rs, 1, 32);
      rs += __shfl_xor(rs, 2, 32);
      rs += __shfl_xor(rs, 4, 32);
      rs += __shfl_xor(rs, 8, 32);
      lrow[r] = lrow[r] * alpha + rs;
#pragma unroll
      for (int c = 0; c < 8; ++c) o[c][r] *= alpha;
      Pw[(hi8 + r) * 32 + nl]      = (bf16)p0;
      Pw[(hi8 + r) * 32 + 16 + nl] = (bf16)p1;
    }

    // O += P x V
    bf16x16 pa = frag_row<bf16>(Pw, 32);
#pragma unroll
    for (int c = 0; c < 8; ++c) {
      bf16x16 vb = frag_col<bf16>(&Vb[cur][c * 16], HDIM);
      o[c] = wmma_bf16(pa, vb, o[c]);
    }
  }

  bf16* op = attn + ((size_t)(b * S_) + q0 + hi8) * QSIZE + h * HDIM + nl;
#pragma unroll
  for (int r = 0; r < 8; ++r) {
    const float inv = 1.0f / lrow[r];
#pragma unroll
    for (int c = 0; c < 8; ++c) op[(size_t)r * QSIZE + c * 16] = (bf16)(o[c][r] * inv);
  }
}

extern "C" void kernel_launch(void* const* d_in, const int* in_sizes, int n_in,
                              void* d_out, int out_size, void* d_ws, size_t ws_size,
                              hipStream_t stream) {
  const int*   positions = (const int*)d_in[0];
  const float* hidden    = (const float*)d_in[1];
  const float* w_qkv     = (const float*)d_in[2];
  const float* w_o       = (const float*)d_in[3];
  float*       out       = (float*)d_out;

  bf16* qkv  = (bf16*)d_ws;                       // [B*S][3072] bf16 (~25 MB)
  bf16* attn = qkv + (size_t)B_ * S_ * QKV_N;     // [B*S][2048] bf16 (~17 MB)

  dim3 g1(QKV_N / 64, (B_ * S_) / 64);
  gemm_bf16_kernel<float, bf16><<<g1, 128, 0, stream>>>(
      hidden, w_qkv, qkv, HIDDEN_, HIDDEN_, QKV_N, QKV_N);

  const int nrope = B_ * S_ * (NHEADS + NKV) * (HDIM / 2);
  rope_kernel<<<(nrope + 255) / 256, 256, 0, stream>>>(qkv, positions);

  flash_attn_kernel<<<dim3(S_ / 64, NHEADS, B_), 128, 0, stream>>>(qkv, attn);

  dim3 g4(HIDDEN_ / 64, (B_ * S_) / 64);
  gemm_bf16_kernel<bf16, float><<<g4, 128, 0, stream>>>(
      attn, w_o, out, QSIZE, QSIZE, HIDDEN_, HIDDEN_);
}
